// HPLFlowPoseNet_43877385896364
// MI455X (gfx1250) — matrix operand order repaired and checked by reading, not compiled, not run on hardware
//
#include <hip/hip_runtime.h>
#include <math.h>

typedef __bf16 bf16;
typedef __attribute__((ext_vector_type(16))) __bf16 v16bf;
typedef __attribute__((ext_vector_type(8)))  float  v8f;

// ---------------------------------------------------------------------------
// WMMA GEMM:  Y[b,o,n] = act( bias[o] + sum_k W[o,k] * Bmat[k,n] )
//   plain  (GATHER=0): KDIM=C,     Bmat[k,n] = X[b,k,n]
//   gather (GATHER=1): KDIM=C*15,  k->(c=k/15, s=k%15),
//                      Bmat[k,n] = X[b,c, gidx[b,s,n]]   (blur_conv fusion)
// One wave computes a 32(M)x16(N) tile via two v_wmma_f32_16x16x32_bf16
// sharing one B tile. Main K loop is branch-free (unconditional loads);
// the K remainder is peeled and handled with clamped loads + value selects.
// blockDim = 128 (4 waves), grid = (N/64, ceil(O/32), B=2).
// ---------------------------------------------------------------------------
template<int ACT, int GATHER>
__global__ void k_gemm(const float* __restrict__ W, const float* __restrict__ bias,
                       const float* __restrict__ X, const int* __restrict__ gidx,
                       float* __restrict__ Y, int O, int C, int N)
{
  const int KDIM = GATHER ? C * 15 : C;
  const int wave = threadIdx.x >> 5;
  const int lane = threadIdx.x & 31;
  const int n0 = (blockIdx.x * 4 + wave) * 16;
  const int m0 = blockIdx.y * 32;
  const int b  = blockIdx.z;
  if (n0 >= N) return;                      // wave-uniform: EXEC stays all-ones

  const float* Xb = X + (size_t)b * C * N;
  float*       Yb = Y + (size_t)b * O * N;

  const int mA = lane & 15;   // A-matrix row within tile
  const int kh = lane >> 4;   // K-half selector (ISA lane split)
  const int nB = lane & 15;   // B-matrix column within tile

  // Clamp A rows: rows >= O read row O-1; their results are never stored.
  int r0 = m0 + mA;       if (r0 > O - 1) r0 = O - 1;
  int r1 = m0 + 16 + mA;  if (r1 > O - 1) r1 = O - 1;
  const float* W0 = W + (size_t)r0 * KDIM;
  const float* W1 = W + (size_t)r1 * KDIM;
  const float* Xcol = Xb + n0 + nB;
  const int*   Gcol = GATHER ? (gidx + (size_t)b * 15 * N + n0 + nB) : (const int*)0;

  v8f acc0 = {}, acc1 = {};
  const int kmain = KDIM & ~31;
  int k0 = 0;
  // ---- branch-free main loop ----
  for (; k0 < kmain; k0 += 32) {
    v16bf a0, a1, bm;
#pragma unroll
    for (int i = 0; i < 16; ++i) {
      int kk = k0 + kh * 8 + ((i < 8) ? i : i + 8);
      a0[i] = (bf16)W0[kk];
      a1[i] = (bf16)W1[kk];
    }
#pragma unroll
    for (int i = 0; i < 16; ++i) {
      int kk = k0 + kh * 8 + ((i < 8) ? i : i + 8);
      float bv;
      if (GATHER) {
        int c = kk / 15;
        int s = kk - c * 15;
        bv = Xb[(size_t)c * N + (size_t)Gcol[(size_t)s * N]];
      } else {
        bv = Xcol[(size_t)kk * N];
      }
      bm[i] = (bf16)bv;
    }
    acc0 = __builtin_amdgcn_wmma_f32_16x16x32_bf16(false, a0, false, bm,
                                                   (short)0, acc0, false, false);
    acc1 = __builtin_amdgcn_wmma_f32_16x16x32_bf16(false, a1, false, bm,
                                                   (short)0, acc1, false, false);
  }
  // ---- peeled K tail: clamped addresses, value-masked (no control flow) ----
  if (k0 < KDIM) {
    v16bf a0, a1, bm;
#pragma unroll
    for (int i = 0; i < 16; ++i) {
      int kk = k0 + kh * 8 + ((i < 8) ? i : i + 8);
      int kc = (kk < KDIM) ? kk : (KDIM - 1);
      float v0 = W0[kc];
      float v1 = W1[kc];
      a0[i] = (bf16)((kk < KDIM) ? v0 : 0.f);
      a1[i] = (bf16)((kk < KDIM) ? v1 : 0.f);
    }
#pragma unroll
    for (int i = 0; i < 16; ++i) {
      int kk = k0 + kh * 8 + ((i < 8) ? i : i + 8);
      int kc = (kk < KDIM) ? kk : (KDIM - 1);
      float bv;
      if (GATHER) {
        int c = kc / 15;
        int s = kc - c * 15;
        bv = Xb[(size_t)c * N + (size_t)Gcol[(size_t)s * N]];
      } else {
        bv = Xcol[(size_t)kc * N];
      }
      bm[i] = (bf16)((kk < KDIM) ? bv : 0.f);
    }
    acc0 = __builtin_amdgcn_wmma_f32_16x16x32_bf16(false, a0, false, bm,
                                                   (short)0, acc0, false, false);
    acc1 = __builtin_amdgcn_wmma_f32_16x16x32_bf16(false, a1, false, bm,
                                                   (short)0, acc1, false, false);
  }
  // D layout: VGPR r -> M = tile_m0 + r + 8*kh ; N = n0 + (lane&15)
  const int col = n0 + nB;
#pragma unroll
  for (int r = 0; r < 8; ++r) {
    int rr = m0 + r + kh * 8;
    if (rr < O) {
      float v = acc0[r] + bias[rr];
      if (ACT) v = (v > 0.f) ? v : 0.1f * v;
      Yb[(size_t)rr * N + col] = v;
    }
  }
#pragma unroll
  for (int r = 0; r < 8; ++r) {
    int rr = m0 + 16 + r + kh * 8;
    if (rr < O) {
      float v = acc1[r] + bias[rr];
      if (ACT) v = (v > 0.f) ? v : 0.1f * v;
      Yb[(size_t)rr * N + col] = v;
    }
  }
}

// ---------------------------------------------------------------------------
// Elementwise / gather / scatter helpers
// ---------------------------------------------------------------------------
__global__ void k_copych(float* __restrict__ dst, int dstC, int coff,
                         const float* __restrict__ src, int C, int N)
{
  int idx = blockIdx.x * blockDim.x + threadIdx.x;
  int total = 2 * C * N;
  if (idx >= total) return;
  int n = idx % N; int t = idx / N; int c = t % C; int b = t / C;
  dst[((size_t)b * dstC + coff + c) * N + n] = src[idx];
}

__global__ void k_splat_add(const float* __restrict__ feat, const float* __restrict__ bary,
                            const int* __restrict__ off, float* __restrict__ lat,
                            int C, int n, int Hs)
{
  int idx = blockIdx.x * blockDim.x + threadIdx.x;
  int total = 2 * C * n;
  if (idx >= total) return;
  int p = idx % n; int t = idx / n; int c = t % C; int b = t / C;
  float f = feat[idx];
#pragma unroll
  for (int j = 0; j < 4; ++j) {
    float w = bary[((size_t)b * 4 + j) * n + p];
    int   o = off [((size_t)b * 4 + j) * n + p];
    atomicAdd(&lat[((size_t)b * C + c) * Hs + o], f * w);
  }
}

__global__ void k_norm_add(const float* __restrict__ bary, const int* __restrict__ off,
                           float* __restrict__ norm, int n, int Hs)
{
  int idx = blockIdx.x * blockDim.x + threadIdx.x;
  if (idx >= 2 * n) return;
  int p = idx % n; int b = idx / n;
#pragma unroll
  for (int j = 0; j < 4; ++j) {
    float w = bary[((size_t)b * 4 + j) * n + p];
    int   o = off [((size_t)b * 4 + j) * n + p];
    atomicAdd(&norm[(size_t)b * Hs + o], w);
  }
}

__global__ void k_splat_div(float* __restrict__ lat, const float* __restrict__ norm,
                            int C, int Hs)
{
  int idx = blockIdx.x * blockDim.x + threadIdx.x;
  int total = 2 * C * Hs;
  if (idx >= total) return;
  int h = idx % Hs; int b = idx / (C * Hs);
  lat[idx] /= fmaxf(norm[(size_t)b * Hs + h], 1e-8f);
}

__global__ void k_corr_raw(const float* __restrict__ o1f, const float* __restrict__ o2f,
                           const int* __restrict__ cidx, float* __restrict__ raw, int Hs)
{
  int idx = blockIdx.x * blockDim.x + threadIdx.x;
  int total = 2 * 15 * Hs;
  if (idx >= total) return;
  int h = idx % Hs; int b = idx / (15 * Hs);
  int g = cidx[idx];
  const float* p1 = o1f + (size_t)b * 64 * Hs + h;
  const float* p2 = o2f + (size_t)b * 64 * Hs + g;
  float s = 0.f;
#pragma unroll 8
  for (int c = 0; c < 64; ++c) s += p1[(size_t)c * Hs] * p2[(size_t)c * Hs];
  raw[idx] = s * (1.f / 64.f);
}

__global__ void k_slice(const float* __restrict__ lat, const float* __restrict__ bary,
                        const int* __restrict__ off, float* __restrict__ out,
                        int C, int n, int Hs)
{
  int idx = blockIdx.x * blockDim.x + threadIdx.x;
  int total = 2 * C * n;
  if (idx >= total) return;
  int p = idx % n; int t = idx / n; int c = t % C; int b = t / C;
  float s = 0.f;
#pragma unroll
  for (int j = 0; j < 4; ++j) {
    float w = bary[((size_t)b * 4 + j) * n + p];
    int   o = off [((size_t)b * 4 + j) * n + p];
    s += lat[((size_t)b * C + c) * Hs + o] * w;
  }
  out[idx] = s;
}

__global__ void k_rowmax(const float* __restrict__ x, float* __restrict__ y, int C, int N)
{
  int idx = blockIdx.x * blockDim.x + threadIdx.x;
  if (idx >= 2 * C) return;
  const float* p = x + (size_t)idx * N;
  float m = p[0];
  for (int n = 1; n < N; ++n) m = fmaxf(m, p[n]);
  y[idx] = m;
}

__global__ void k_fc(const float* __restrict__ W, const float* __restrict__ bias,
                     const float* __restrict__ x1, const float* __restrict__ x2,
                     float* __restrict__ y, int O, int C1, int C2, int act)
{
  int idx = blockIdx.x * blockDim.x + threadIdx.x;
  if (idx >= 2 * O) return;
  int o = idx % O; int b = idx / O;
  const float* Wr = W + (size_t)o * (C1 + C2);
  float s = bias[o];
  for (int c = 0; c < C1; ++c) s += Wr[c] * x1[(size_t)b * C1 + c];
  for (int c = 0; c < C2; ++c) s += Wr[C1 + c] * x2[(size_t)b * C2 + c];
  if (act && s < 0.f) s *= 0.1f;
  y[idx] = s;
}

// R = Rz @ Ry @ Rx from ang = out6[b][3..5]; pc = R @ pc + trans (in-place)
__global__ void k_pose_apply(const float* __restrict__ out6, float* __restrict__ pc, int N)
{
  int idx = blockIdx.x * blockDim.x + threadIdx.x;
  if (idx >= 2 * N) return;
  int n = idx % N; int b = idx / N;
  float tx = out6[b * 6 + 0], ty = out6[b * 6 + 1], tz = out6[b * 6 + 2];
  float ax = out6[b * 6 + 3], ay = out6[b * 6 + 4], az = out6[b * 6 + 5];
  float cx = cosf(ax), sx = sinf(ax);
  float cy = cosf(ay), sy = sinf(ay);
  float cz = cosf(az), sz = sinf(az);
  float r00 = cz * cy, r01 = cz * sy * sx - sz * cx, r02 = cz * sy * cx + sz * sx;
  float r10 = sz * cy, r11 = sz * sy * sx + cz * cx, r12 = sz * sy * cx - cz * sx;
  float r20 = -sy,     r21 = cy * sx,                r22 = cy * cx;
  float* p = pc + (size_t)b * 3 * N + n;
  float X = p[0], Y = p[N], Z = p[2 * N];
  p[0]     = r00 * X + r01 * Y + r02 * Z + tx;
  p[N]     = r10 * X + r11 * Y + r12 * Z + ty;
  p[2 * N] = r20 * X + r21 * Y + r22 * Z + tz;
}

__global__ void k_final(float* __restrict__ out, const float* __restrict__ pc1cm,
                        const float* __restrict__ pc1, int total)
{
  int idx = blockIdx.x * blockDim.x + threadIdx.x;
  if (idx >= total) return;
  out[idx] += pc1cm[idx] - pc1[idx];
}

// ---------------------------------------------------------------------------
// Orchestration
// ---------------------------------------------------------------------------
static const int Hh[7]   = {4096, 2048, 1024, 512, 256, 128, 64};
static const int NP[7]   = {8192, 4096, 2048, 1024, 512, 256, 128};
static const int UP_I[7] = {580, 324, 388, 260, 260, 260, 128};
static const int UP_O[7] = {1024, 512, 256, 256, 128, 128, 128};

extern "C" void kernel_launch(void* const* d_in, const int* in_sizes, int n_in,
                              void* d_out, int out_size, void* d_ws, size_t ws_size,
                              hipStream_t stream)
{
  (void)in_sizes; (void)n_in; (void)out_size; (void)ws_size;
  auto F = [&](int i) { return (const float*)d_in[i]; };
  auto I = [&](int i) { return (const int*)d_in[i]; };

  // Input index map (setup_inputs dict insertion order):
  //  0 pc1, 1 pc2
  //  conv1: 2..7 (W,b)x3
  //  bcn_down i: 8+4i .. 11+4i (W1,b1,W2,b2)
  //  bcn_up lev: 36+4lev .. 39+4lev
  //  corr lev(2..6): 64+8(lev-2): Wc1,bc1,Wc2,bc2,Wm1,bm1,Wm2,bm2
  //  conv2 104/105, conv3 106/107, conv4 108/109
  //  pose enc 110..115, p4 116/117, p5 118/119
  //  lattice pc1: bary 120+i, off 127+i, blur 134+i, el 141+i
  //  lattice pc2: bary 148+i, off 155+i, blur 162+i, el 169+i
  //  el_up 176+i (i<6), corr_idx 182+(lev-2)
  const float* pc1 = F(0);
  const float* pc2 = F(1);

  // ---- workspace bump allocation (floats) ----
  float* ws = (float*)d_ws;
  size_t woff = 0;
  auto alloc = [&](size_t nf) { float* p = ws + woff; woff += nf; return p; };
  float* pc1cm = alloc(2 * 3 * 8192);
  float* feat1 = alloc((size_t)2 * 64 * 8192);
  float* feat2 = alloc((size_t)2 * 64 * 8192);
  float* o1l[7]; for (int i = 0; i < 7; ++i) o1l[i] = alloc((size_t)2 * 64 * Hh[i]);
  float* o2l[7]; for (int i = 0; i < 7; ++i) o2l[i] = alloc((size_t)2 * 64 * Hh[i]);
  float* cl[5];  for (int i = 0; i < 5; ++i) cl[i]  = alloc((size_t)2 * 64 * Hh[i + 2]);
  float* Ubuf = alloc((size_t)2 * 1024 * 8192);
  float* H1   = alloc((size_t)2 * 1024 * 8192);
  float* Ta   = alloc((size_t)2 * 1024 * 4096);
  float* Tb   = alloc((size_t)2 * 1024 * 4096);
  float* Xcat = alloc((size_t)2 * 580 * 4096);
  float* Latb = alloc((size_t)2 * 68 * 4096);
  float* Xbuf = alloc((size_t)2 * 68 * 8192);
  float* Normb = alloc(2 * 4096);
  float* Raw   = alloc(2 * 15 * 1024);
  float* Y32a  = alloc(2 * 32 * 1024);
  float* Y32b  = alloc(2 * 32 * 1024);
  float* Lat64 = alloc(2 * 64 * 1024);
  float* Ycat  = alloc(2 * 96 * 1024);
  float* F1s  = alloc(512);
  float* F2s  = alloc(512);
  float* Hsml = alloc(512);
  float* Out6 = alloc(16);

  // ---- launch helpers ----
  auto gemm = [&](const float* W, const float* bi, const float* X, float* Y,
                  int O, int C, int N, int act) {
    dim3 g(N / 64, (O + 31) / 32, 2), blk(128, 1, 1);
    if (act) k_gemm<1, 0><<<g, blk, 0, stream>>>(W, bi, X, (const int*)0, Y, O, C, N);
    else     k_gemm<0, 0><<<g, blk, 0, stream>>>(W, bi, X, (const int*)0, Y, O, C, N);
  };
  auto ggemm = [&](const float* W, const float* bi, const float* X, const int* gx,
                   float* Y, int O, int C, int N) {
    dim3 g(N / 64, (O + 31) / 32, 2), blk(128, 1, 1);
    k_gemm<1, 1><<<g, blk, 0, stream>>>(W, bi, X, gx, Y, O, C, N);
  };
  auto copych = [&](float* dst, int dstC, int coff, const float* src, int C, int N) {
    int tot = 2 * C * N;
    k_copych<<<dim3((tot + 255) / 256), dim3(256), 0, stream>>>(dst, dstC, coff, src, C, N);
  };
  auto splat = [&](const float* feat, const float* ba, const int* of,
                   float* lat, float* nrm, int C, int n, int Hs) {
    hipMemsetAsync(lat, 0, (size_t)2 * C * Hs * sizeof(float), stream);
    hipMemsetAsync(nrm, 0, (size_t)2 * Hs * sizeof(float), stream);
    int t1 = 2 * C * n;
    k_splat_add<<<dim3((t1 + 255) / 256), dim3(256), 0, stream>>>(feat, ba, of, lat, C, n, Hs);
    int t2 = 2 * n;
    k_norm_add<<<dim3((t2 + 255) / 256), dim3(256), 0, stream>>>(ba, of, nrm, n, Hs);
    int t3 = 2 * C * Hs;
    k_splat_div<<<dim3((t3 + 255) / 256), dim3(256), 0, stream>>>(lat, nrm, C, Hs);
  };

  // ================= pose iterations =================
  hipMemcpyAsync(pc1cm, pc1, sizeof(float) * 2 * 3 * 8192, hipMemcpyDeviceToDevice, stream);
  // f2 (depends only on pc2 -> compute once, identical to recomputation)
  gemm(F(110), F(111), pc2, Ta, 64, 3, 8192, 1);
  gemm(F(112), F(113), Ta, Tb, 128, 64, 8192, 1);
  gemm(F(114), F(115), Tb, Ta, 256, 128, 8192, 1);
  k_rowmax<<<dim3(2), dim3(256), 0, stream>>>(Ta, F2s, 256, 8192);
  for (int it = 0; it < 5; ++it) {
    gemm(F(110), F(111), pc1cm, Ta, 64, 3, 8192, 1);
    gemm(F(112), F(113), Ta, Tb, 128, 64, 8192, 1);
    gemm(F(114), F(115), Tb, Ta, 256, 128, 8192, 1);
    k_rowmax<<<dim3(2), dim3(256), 0, stream>>>(Ta, F1s, 256, 8192);
    k_fc<<<dim3(2), dim3(256), 0, stream>>>(F(116), F(117), F1s, F2s, Hsml, 256, 256, 256, 1);
    k_fc<<<dim3(1), dim3(32), 0, stream>>>(F(118), F(119), Hsml, (const float*)0, Out6, 6, 256, 0, 0);
    k_pose_apply<<<dim3((16384 + 255) / 256), dim3(256), 0, stream>>>(Out6, pc1cm, 8192);
  }

  // ================= conv1 features =================
  gemm(F(2), F(3), pc1cm, Ta, 32, 3, 8192, 1);
  gemm(F(4), F(5), Ta, Tb, 32, 32, 8192, 1);
  gemm(F(6), F(7), Tb, feat1, 64, 32, 8192, 1);
  gemm(F(2), F(3), pc2, Ta, 32, 3, 8192, 1);
  gemm(F(4), F(5), Ta, Tb, 32, 32, 8192, 1);
  gemm(F(6), F(7), Tb, feat2, 64, 32, 8192, 1);

  // ================= down path (both clouds) =================
  auto down = [&](const float* feat0, int latbase, float* const* olev) {
    const float* f = feat0;
    for (int i = 0; i < 7; ++i) {
      int n = NP[i], Hs = Hh[i];
      copych(Xbuf, 68, 0, F(latbase + 21 + i), 4, n);   // el
      copych(Xbuf, 68, 4, f, 64, n);
      splat(Xbuf, F(latbase + i), I(latbase + 7 + i), Latb, Normb, 68, n, Hs);
      ggemm(F(8 + 4 * i), F(9 + 4 * i), Latb, I(latbase + 14 + i), Ta, 64, 68, Hs);
      gemm(F(10 + 4 * i), F(11 + 4 * i), Ta, olev[i], 64, 64, Hs, 1);
      f = olev[i];
    }
  };
  down(feat1, 120, o1l);
  down(feat2, 148, o2l);

  // ================= correlation pyramid =================
  auto corr = [&](int lev, const float* prev, float* out) {
    int Hs = Hh[lev];
    int base = 64 + 8 * (lev - 2);
    int t = 2 * 15 * Hs;
    k_corr_raw<<<dim3((t + 255) / 256), dim3(256), 0, stream>>>(
        o1l[lev], o2l[lev], I(182 + lev - 2), Raw, Hs);
    gemm(F(base + 0), F(base + 1), Raw, Y32a, 32, 15, Hs, 1);
    gemm(F(base + 2), F(base + 3), Y32a, Y32b, 32, 32, Hs, 1);
    const float* min_in; int minC;
    if (prev) {
      splat(prev, F(120 + lev), I(127 + lev), Lat64, Normb, 64, NP[lev], Hs);
      copych(Ycat, 96, 0, Y32b, 32, Hs);
      copych(Ycat, 96, 32, Lat64, 64, Hs);
      min_in = Ycat; minC = 96;
    } else {
      min_in = Y32b; minC = 32;
    }
    gemm(F(base + 4), F(base + 5), min_in, Lat64, 64, minC, Hs, 1);
    gemm(F(base + 6), F(base + 7), Lat64, out, 64, 64, Hs, 1);
  };
  corr(2, (const float*)0, cl[0]);
  corr(3, cl[0], cl[1]);
  corr(4, cl[1], cl[2]);
  corr(5, cl[2], cl[3]);
  corr(6, cl[3], cl[4]);

  // ================= up path =================
  // level 6: concat([o1[6], c5])
  {
    int Hs = Hh[6];
    copych(Xcat, 128, 0, o1l[6], 64, Hs);
    copych(Xcat, 128, 64, cl[4], 64, Hs);
    int base = 36 + 4 * 6;
    ggemm(F(base), F(base + 1), Xcat, I(134 + 6), Ta, 128, 128, Hs);
    gemm(F(base + 2), F(base + 3), Ta, Tb, 128, 128, Hs, 1);
    int tot = 2 * 128 * NP[6];
    k_slice<<<dim3((tot + 255) / 256), dim3(256), 0, stream>>>(
        Tb, F(120 + 6), I(127 + 6), Ubuf, 128, NP[6], Hs);
  }
  int uch = 128;
  for (int lev = 5; lev >= 0; --lev) {
    int Hs = Hh[lev];
    int ich = UP_I[lev], och = UP_O[lev];
    int co = 0;
    copych(Xcat, ich, co, F(176 + lev), 4, Hs); co += 4;     // el_up
    copych(Xcat, ich, co, o1l[lev], 64, Hs);    co += 64;
    if (lev >= 2) { copych(Xcat, ich, co, cl[lev - 2], 64, Hs); co += 64; }
    copych(Xcat, ich, co, Ubuf, uch, Hs);       co += uch;   // co == ich
    int base = 36 + 4 * lev;
    ggemm(F(base), F(base + 1), Xcat, I(134 + lev), Ta, och, ich, Hs);
    gemm(F(base + 2), F(base + 3), Ta, Tb, och, och, Hs, 1);
    int tot = 2 * och * NP[lev];
    k_slice<<<dim3((tot + 255) / 256), dim3(256), 0, stream>>>(
        Tb, F(120 + lev), I(127 + lev), Ubuf, och, NP[lev], Hs);
    uch = och;
  }

  // ================= head =================
  gemm(F(104), F(105), Ubuf, H1, 1024, 1024, 8192, 1);
  gemm(F(106), F(107), H1, Ta, 512, 1024, 8192, 1);
  gemm(F(108), F(109), Ta, (float*)d_out, 3, 512, 8192, 0);
  k_final<<<dim3((49152 + 255) / 256), dim3(256), 0, stream>>>(
      (float*)d_out, pc1cm, pc1, 49152);
}
